// QLSTMHybrid_65481071404101
// MI455X (gfx1250) — matrix operand order repaired and checked by reading, not compiled, not run on hardware
//
#include <hip/hip_runtime.h>
#include <hip/hip_bf16.h>

// ---- problem dims (match reference) ----
#define SEQ    512
#define BATCH  64
#define IN_DIM 512
#define HID    1024
#define CDIM   (IN_DIM + HID)   // 1536
#define NBLK   ((HID / 16) * (BATCH / 16))   // 256 persistent blocks

typedef __bf16 v16bf __attribute__((ext_vector_type(16)));
typedef __bf16 v8bf  __attribute__((ext_vector_type(8)));
typedef float  v8f   __attribute__((ext_vector_type(8)));

static __device__ __forceinline__ __bf16 f32_to_bf16(float f) {
  union { float f; unsigned u; } v; v.f = f;
  unsigned r = v.u + 0x7FFFu + ((v.u >> 16) & 1u);   // round-to-nearest-even
  unsigned short h = (unsigned short)(r >> 16);
  __bf16 o; __builtin_memcpy(&o, &h, 2); return o;
}

static __device__ __forceinline__ v16bf ld_pair(const __bf16* p0, const __bf16* p1) {
  union { v16bf v; v8bf h[2]; } u;
  u.h[0] = *reinterpret_cast<const v8bf*>(p0);   // b128 load
  u.h[1] = *reinterpret_cast<const v8bf*>(p1);   // b128 load
  return u.v;
}

static __device__ __forceinline__ float sigmoidf_(float x) {
  return 1.0f / (1.0f + __expf(-x));
}
static __device__ __forceinline__ float tanhf_(float x) {
  float e = __expf(2.0f * x);
  return (e - 1.0f) / (e + 1.0f);
}

// Device-wide sense-reversing barrier (agent scope). All NBLK blocks are resident.
static __device__ __forceinline__ void grid_barrier(unsigned* count, unsigned* gen) {
  __syncthreads();
  if (threadIdx.x == 0) {
    unsigned g = __hip_atomic_load(gen, __ATOMIC_RELAXED, __HIP_MEMORY_SCOPE_AGENT);
    unsigned prev = __hip_atomic_fetch_add(count, 1u, __ATOMIC_ACQ_REL, __HIP_MEMORY_SCOPE_AGENT);
    if (prev == NBLK - 1u) {
      __hip_atomic_store(count, 0u, __ATOMIC_RELAXED, __HIP_MEMORY_SCOPE_AGENT);
      __hip_atomic_store(gen, g + 1u, __ATOMIC_RELEASE, __HIP_MEMORY_SCOPE_AGENT);
    } else {
      while (__hip_atomic_load(gen, __ATOMIC_ACQUIRE, __HIP_MEMORY_SCOPE_AGENT) == g) {
        __builtin_amdgcn_s_sleep(2);
      }
    }
  }
  __syncthreads();
}

// ---- one-time conversions ----

// W (1536,1024) f32 row-major -> WT[g][n][k] bf16, LDS-tiled transpose
__global__ void wt_transpose_kernel(const float* __restrict__ Wf, const float* __restrict__ Wi,
                                    const float* __restrict__ Wg, const float* __restrict__ Wo,
                                    __bf16* __restrict__ WT) {
  __shared__ __bf16 tile[32][33];
  const int g = blockIdx.z;
  const float* W = (g == 0) ? Wf : (g == 1) ? Wi : (g == 2) ? Wg : Wo;
  const int k0 = blockIdx.y * 32;
  const int n0 = blockIdx.x * 32;
  const int tx = threadIdx.x, ty = threadIdx.y;   // (32, 8)
#pragma unroll
  for (int i = 0; i < 32; i += 8)
    tile[ty + i][tx] = f32_to_bf16(W[(size_t)(k0 + ty + i) * HID + (n0 + tx)]);
  __syncthreads();
#pragma unroll
  for (int i = 0; i < 32; i += 8)
    WT[(size_t)g * HID * CDIM + (size_t)(n0 + ty + i) * CDIM + (k0 + tx)] = tile[tx][ty + i];
}

__global__ void x_convert_kernel(const float* __restrict__ x, __bf16* __restrict__ xbf) {
  const size_t i = ((size_t)blockIdx.x * blockDim.x + threadIdx.x) * 4;
  float4 v = *reinterpret_cast<const float4*>(x + i);
  __bf16 t[4] = { f32_to_bf16(v.x), f32_to_bf16(v.y), f32_to_bf16(v.z), f32_to_bf16(v.w) };
  uint2 bits; __builtin_memcpy(&bits, t, 8);
  *reinterpret_cast<uint2*>(xbf + i) = bits;
}

__global__ void init_state_kernel(float* __restrict__ c, __bf16* __restrict__ h0,
                                  unsigned* __restrict__ bar) {
  const int i = blockIdx.x * blockDim.x + threadIdx.x;  // BATCH*HID threads
  c[i] = 0.0f;
  unsigned short z = 0; __builtin_memcpy(&h0[i], &z, 2);
  if (i < 2) bar[i] = 0u;   // barrier count + generation
}

// ---- persistent full-sequence scan ----
// grid (HID/16, BATCH/16) = 256 blocks, 128 threads = 4 waves, wave = one gate (f,i,g,o).
// Per step each wave: 16x16 f32 tile, K=1536 in steps of 32 -> 48 v_wmma_f32_16x16x32_bf16.
// One device-wide barrier per timestep replaces 512 kernel launches.
__global__ __launch_bounds__(128)
void lstm_scan_kernel(const __bf16* __restrict__ xbf,   // (SEQ, BATCH, IN_DIM) bf16
                      const __bf16* __restrict__ WT,    // (4, HID, CDIM) bf16
                      const float*  __restrict__ b_f,
                      const float*  __restrict__ b_i,
                      const float*  __restrict__ b_g,
                      const float*  __restrict__ b_o,
                      float*  __restrict__ c_state,     // (BATCH, HID) f32, block-owned
                      __bf16* __restrict__ hbuf0,       // ping-pong h (BATCH, HID) bf16
                      __bf16* __restrict__ hbuf1,
                      float*  __restrict__ out,         // (SEQ, BATCH, HID) f32
                      float*  __restrict__ hx_out,
                      float*  __restrict__ cx_out,
                      unsigned* __restrict__ bar) {
  const int lane  = threadIdx.x & 31;
  const int gate  = threadIdx.x >> 5;
  const int nrow  = lane & 15;      // A: row-in-tile, B: col-in-tile, C/D: N
  const int khalf = lane >> 4;
  const int m0 = blockIdx.y << 4;
  const int n0 = blockIdx.x << 4;

  const __bf16* wp = WT + (size_t)gate * HID * CDIM + (size_t)(n0 + nrow) * CDIM;
  const float* bp = (gate == 0) ? b_f : (gate == 1) ? b_i : (gate == 2) ? b_g : b_o;
  const float bias = bp[n0 + nrow];           // C/D N == lane&15 for all 8 acc regs

  const size_t arow = (size_t)(m0 + nrow);
  __shared__ float gbuf[4][256];

  for (int t = 0; t < SEQ; ++t) {
    const __bf16* ax = xbf + ((size_t)t * BATCH + arow) * IN_DIM;
    const __bf16* ah = ((t & 1) ? hbuf1 : hbuf0) + arow * HID;
    __bf16* h_out    = (t & 1) ? hbuf0 : hbuf1;
    float*  out_t    = out + (size_t)t * BATCH * HID;

    v8f acc;
#pragma unroll
    for (int e = 0; e < 8; ++e) acc[e] = bias;

    // K over the x part of comb (512)
#pragma unroll 4
    for (int k0 = 0; k0 < IN_DIM; k0 += 32) {
      v16bf a = ld_pair(ax + k0 + khalf * 8,  ax + k0 + khalf * 8 + 16);
      v16bf b = ld_pair(wp + k0 + khalf * 8,  wp + k0 + khalf * 8 + 16);
      acc = __builtin_amdgcn_wmma_f32_16x16x32_bf16(false, a, false, b, (short)0, acc, false, false);
    }
    // K over the h part of comb (1024); weights offset by IN_DIM in k
#pragma unroll 4
    for (int k0 = 0; k0 < HID; k0 += 32) {
      v16bf a = ld_pair(ah + k0 + khalf * 8, ah + k0 + khalf * 8 + 16);
      v16bf b = ld_pair(wp + IN_DIM + k0 + khalf * 8, wp + IN_DIM + k0 + khalf * 8 + 16);
      acc = __builtin_amdgcn_wmma_f32_16x16x32_bf16(false, a, false, b, (short)0, acc, false, false);
    }

    // exchange the four 16x16 gate tiles through LDS, then fused LSTM update
#pragma unroll
    for (int e = 0; e < 8; ++e)
      gbuf[gate][(e + khalf * 8) * 16 + nrow] = acc[e];   // M = e + 8*khalf, N = nrow
    __syncthreads();

    for (int e = threadIdx.x; e < 256; e += 128) {
      const float fv = sigmoidf_(gbuf[0][e]);
      const float iv = sigmoidf_(gbuf[1][e]);
      const float gv = tanhf_(gbuf[2][e]);
      const float ov = sigmoidf_(gbuf[3][e]);
      const int   bb = m0 + (e >> 4);
      const int   nn = n0 + (e & 15);
      const size_t idx = (size_t)bb * HID + nn;
      const float cv = fv * c_state[idx] + iv * gv;
      const float hv = ov * tanhf_(cv);
      c_state[idx] = cv;
      h_out[idx]   = f32_to_bf16(hv);
      out_t[idx]   = hv;
      if (t == SEQ - 1) { hx_out[idx] = hv; cx_out[idx] = cv; }
    }

    // all blocks must finish step t (h_out written, h_in reads done) before t+1
    grid_barrier(bar + 0, bar + 1);
  }
}

extern "C" void kernel_launch(void* const* d_in, const int* in_sizes, int n_in,
                              void* d_out, int out_size, void* d_ws, size_t ws_size,
                              hipStream_t stream) {
  (void)in_sizes; (void)n_in; (void)out_size; (void)ws_size;

  const float* x   = (const float*)d_in[0];
  const float* Wf  = (const float*)d_in[1];
  const float* b_f = (const float*)d_in[2];
  const float* Wi  = (const float*)d_in[3];
  const float* b_i = (const float*)d_in[4];
  const float* Wg  = (const float*)d_in[5];
  const float* b_g = (const float*)d_in[6];
  const float* Wo  = (const float*)d_in[7];
  const float* b_o = (const float*)d_in[8];

  char* ws = (char*)d_ws;
  size_t off = 0;
  __bf16*   WT  = (__bf16*)(ws + off);   off += (size_t)4 * HID * CDIM * 2;        // 12.6 MB
  __bf16*   xbf = (__bf16*)(ws + off);   off += (size_t)SEQ * BATCH * IN_DIM * 2;  // 32 MB
  __bf16*   h0  = (__bf16*)(ws + off);   off += (size_t)BATCH * HID * 2;           // 128 KB
  __bf16*   h1  = (__bf16*)(ws + off);   off += (size_t)BATCH * HID * 2;           // 128 KB
  float*    cst = (float*)  (ws + off);  off += (size_t)BATCH * HID * 4;           // 256 KB
  unsigned* bar = (unsigned*)(ws + off); off += 256;                               // barrier words

  float* out = (float*)d_out;
  float* hx  = out + (size_t)SEQ * BATCH * HID;
  float* cx  = hx  + (size_t)BATCH * HID;

  wt_transpose_kernel<<<dim3(HID / 32, CDIM / 32, 4), dim3(32, 8), 0, stream>>>(Wf, Wi, Wg, Wo, WT);
  x_convert_kernel<<<(SEQ * BATCH * IN_DIM / 4) / 256, 256, 0, stream>>>(x, xbf);
  init_state_kernel<<<(BATCH * HID) / 256, 256, 0, stream>>>(cst, h0, bar);

  lstm_scan_kernel<<<dim3(HID / 16, BATCH / 16), 128, 0, stream>>>(
      xbf, WT, b_f, b_i, b_g, b_o, cst, h0, h1, out, hx, cx, bar);
}
